// SoftNPLoss_17892833755270
// MI455X (gfx1250) — matrix operand order repaired and checked by reading, not compiled, not run on hardware
//
#include <hip/hip_runtime.h>

// ---------------------------------------------------------------------------
// SoftNP loss, fused, for MI455X (gfx1250, wave32).
//   B=2048 rows, D=32 dims, K=30 pre-kNN, N=8192 global ids.
//   Gram tiles via V_WMMA_F32_16X16X32_F16 (D=32 == one WMMA K exactly).
//   Mask via inverted-index bitmap precompute (61K scatter ops instead of
//   126M per-pair compares). Raw v_sqrt_f32 / v_exp_f32 transcendentals.
//   Deterministic fixed-order float reductions.
// ---------------------------------------------------------------------------

typedef __attribute__((ext_vector_type(16))) _Float16 v16h;
typedef __attribute__((ext_vector_type(8)))  _Float16 v8h;
typedef __attribute__((ext_vector_type(8)))  float    v8f;

static constexpr int ZD   = 32;     // feature dim == WMMA K
static constexpr int KNN  = 30;     // pre-kNN per sample
static constexpr float EPSV   = 1e-8f;
static constexpr float NLOG2E = -1.44269504088896340736f;  // -log2(e)

// --- prep: f32 z -> f16 z, plus f32 squared row norms --------------------
__global__ void softnp_prep(const float* __restrict__ z,
                            _Float16* __restrict__ zh,
                            float* __restrict__ sqn, int B) {
  int row = blockIdx.x * blockDim.x + threadIdx.x;
  if (row >= B) return;
  const float* zr = z + row * ZD;
  _Float16*    zo = zh + row * ZD;
  float s = 0.f;
#pragma unroll
  for (int d = 0; d < ZD; ++d) {
    float v = zr[d];
    s += v * v;
    zo[d] = (_Float16)v;
  }
  sqn[row] = s;
}

// --- mask step 0: init inverse-map heads (-1) and zero the bitmap --------
__global__ void softnp_mask_init(int* __restrict__ head, unsigned* __restrict__ maskT,
                                 int N, int nwords) {
  int t = blockIdx.x * blockDim.x + threadIdx.x;
  if (t < N) head[t] = -1;
  if (t < nwords) maskT[t] = 0u;
}

// --- mask step 1: chain columns by global id: head[g] -> j -> next[j] ----
__global__ void softnp_mask_chain(const int* __restrict__ samp,
                                  int* __restrict__ head, int* __restrict__ nxt,
                                  int B) {
  int j = blockIdx.x * blockDim.x + threadIdx.x;
  if (j >= B) return;
  nxt[j] = atomicExch(&head[samp[j]], j);   // chain order irrelevant: bits OR'd
}

// --- mask step 2: scatter bits. maskT[j][i]: word j*(B/32)+(i>>5) --------
__global__ void softnp_mask_scatter(const int* __restrict__ pre_knn,
                                    const int* __restrict__ samp,
                                    const int* __restrict__ head,
                                    const int* __restrict__ nxt,
                                    unsigned* __restrict__ maskT, int B) {
  int t = blockIdx.x * blockDim.x + threadIdx.x;
  if (t >= B * KNN) return;
  int i = t / KNN, k = t - i * KNN;
  int g = pre_knn[samp[i] * KNN + k];       // global id in row i's pre-kNN set
  int wstride = B >> 5;
  for (int j = head[g]; j != -1; j = nxt[j])
    atomicOr(&maskT[j * wstride + (i >> 5)], 1u << (i & 31));
}

// --- main: one block per 16-row i-tile; 8 waves stride over j-tiles ------
__global__ __launch_bounds__(256)
void softnp_main(const _Float16* __restrict__ zh,
                 const float*    __restrict__ sqn,
                 const unsigned* __restrict__ maskT,
                 float* __restrict__ lossv,
                 float* __restrict__ validv, int B) {
  __shared__ float s_sqni[16];        // ||z_i||^2 for the i-tile
  __shared__ float s_pnum[8][16];     // per-wave partial masked mass
  __shared__ float s_pden[8][16];     // per-wave partial denominator
  __shared__ int   s_pval[8][16];     // per-wave "any masked" flag

  const int tid  = threadIdx.x;
  const int lane = tid & 31;
  const int wave = tid >> 5;
  const int hgrp = lane >> 4;         // half-wave: 0 or 1
  const int n    = lane & 15;         // column within 16-wide tile
  const int i0   = blockIdx.x * 16;
  const int wstride = B >> 5;
  const int wsh  = i0 & 31;           // 0 or 16: where the 16 i-bits sit

  if (tid < 16) s_sqni[tid] = sqn[i0 + tid];
  __syncthreads();

  // A fragment (i-tile, 16x32 f16): lane = row i0+n.
  // Low half-wave holds K {0..7,16..23}; high half holds K {8..15,24..31}.
  const _Float16* za = zh + (i0 + n) * ZD;
  v8h a0 = *(const v8h*)(za + hgrp * 8);
  v8h a1 = *(const v8h*)(za + 16 + hgrp * 8);
  v16h afrag = __builtin_shufflevector(a0, a1,
      0, 1, 2, 3, 4, 5, 6, 7, 8, 9, 10, 11, 12, 13, 14, 15);

  float accn[8], accd[8];
  int   accv[8];
#pragma unroll
  for (int r = 0; r < 8; ++r) { accn[r] = 0.f; accd[r] = 0.f; accv[r] = 0; }

  const int ntiles = B >> 4;
  for (int jt = wave; jt < ntiles; jt += 8) {
    const int j0   = jt * 16;
    const int colj = j0 + n;

    // B fragment (32x16 f16): lane n = column = z-row j0+n, same K order as A:
    // low half K {0..7,16..23}, high half K {8..15,24..31}. Two 16B loads.
    const _Float16* zb = zh + colj * ZD;
    v8h b0 = *(const v8h*)(zb + hgrp * 8);
    v8h b1 = *(const v8h*)(zb + 16 + hgrp * 8);
    v16h bfrag = __builtin_shufflevector(b0, b1,
        0, 1, 2, 3, 4, 5, 6, 7, 8, 9, 10, 11, 12, 13, 14, 15);

    v8f cz = {};
    v8f g = __builtin_amdgcn_wmma_f32_16x16x32_f16(
        /*neg_a=*/false, afrag, /*neg_b=*/false, bfrag,
        /*c_mod=*/(short)0, cz, /*reuse_a=*/false, /*reuse_b=*/false);

    const float    sqj  = sqn[colj];
    // 16 mask bits for (rows i0..i0+15, column colj)
    const unsigned mb   = maskT[colj * wstride + (i0 >> 5)] >> wsh;

#pragma unroll
    for (int r = 0; r < 8; ++r) {
      const int mrow = hgrp * 8 + r;      // C layout: VGPR r -> row 8*half+r
      const int rowi = i0 + mrow;
      float d2 = s_sqni[mrow] + sqj - 2.0f * g[r];
      d2 = fmaxf(d2, 0.0f);
      // exp(-dist): raw TRANS ops (v_sqrt_f32, v_exp_f32); co-execute w/ XDL.
      float dist = __builtin_amdgcn_sqrtf(d2);
      float e = __builtin_amdgcn_exp2f(NLOG2E * dist);   // TEMPERATURE = 1
      const bool diag = (rowi == colj);
      if (diag) e = 0.0f;                 // sim = -inf on diagonal
      const bool m = ((mb >> mrow) & 1u) && !diag;

      accd[r] += e;
      if (m) { accn[r] += e; accv[r] = 1; }
    }
  }

  // Reduce across the 16 lanes of each half-wave (xor 1,2,4,8 stays in-half).
#pragma unroll
  for (int r = 0; r < 8; ++r) {
    for (int off = 1; off < 16; off <<= 1) {
      accn[r] += __shfl_xor(accn[r], off);
      accd[r] += __shfl_xor(accd[r], off);
      accv[r] |= __shfl_xor(accv[r], off);
    }
  }
  if (n == 0) {                           // lanes 0 and 16 publish rows 0-7 / 8-15
#pragma unroll
    for (int r = 0; r < 8; ++r) {
      s_pnum[wave][hgrp * 8 + r] = accn[r];
      s_pden[wave][hgrp * 8 + r] = accd[r];
      s_pval[wave][hgrp * 8 + r] = accv[r];
    }
  }
  __syncthreads();

  // Fixed-order combine of the 8 wave partials -> deterministic.
  if (tid < 16) {
    float num = 0.f, den = 0.f; int vl = 0;
#pragma unroll
    for (int w = 0; w < 8; ++w) {
      num += s_pnum[w][tid];
      den += s_pden[w][tid];
      vl  |= s_pval[w][tid];
    }
    const float s = num / den;
    lossv[i0 + tid]  = vl ? -__logf(s + EPSV) : 0.0f;
    validv[i0 + tid] = vl ? 1.0f : 0.0f;
  }
}

// --- finalize: deterministic fixed-order tree reduction ------------------
__global__ void softnp_finalize(const float* __restrict__ lossv,
                                const float* __restrict__ validv,
                                float* __restrict__ out, int B) {
  __shared__ float ssum[256];
  __shared__ float scnt[256];
  const int tid = threadIdx.x;
  float s = 0.f, c = 0.f;
  for (int i = tid; i < B; i += 256) { s += lossv[i]; c += validv[i]; }
  ssum[tid] = s; scnt[tid] = c;
  __syncthreads();
  for (int off = 128; off > 0; off >>= 1) {
    if (tid < off) { ssum[tid] += ssum[tid + off]; scnt[tid] += scnt[tid + off]; }
    __syncthreads();
  }
  if (tid == 0) out[0] = (scnt[0] > 0.f) ? (ssum[0] / scnt[0]) : 0.0f;
}

extern "C" void kernel_launch(void* const* d_in, const int* in_sizes, int n_in,
                              void* d_out, int out_size, void* d_ws, size_t ws_size,
                              hipStream_t stream) {
  const float* z    = (const float*)d_in[0];   // (B, 32) f32
  const int*   knn  = (const int*)d_in[1];     // (N, 30) i32
  const int*   samp = (const int*)d_in[2];     // (B,)    i32
  const int B = in_sizes[2];                   // 2048
  const int N = in_sizes[1] / KNN;             // 8192
  const int nwords = B * (B >> 5);             // bitmap words

  char* ws = (char*)d_ws;
  _Float16* zh    = (_Float16*)ws;                         // B*32 f16
  float*    sqn   = (float*)(ws + (size_t)B * ZD * 2);     // B f32
  float*    lossv = sqn + B;                               // B f32
  float*    valv  = lossv + B;                             // B f32
  int*      head  = (int*)(valv + B);                      // N i32
  int*      nxt   = head + N;                              // B i32
  unsigned* maskT = (unsigned*)(nxt + B);                  // B*B/32 u32 (B*B bits)

  softnp_prep<<<(B + 255) / 256, 256, 0, stream>>>(z, zh, sqn, B);

  int initn = (N > nwords ? N : nwords);
  softnp_mask_init<<<(initn + 255) / 256, 256, 0, stream>>>(head, maskT, N, nwords);
  softnp_mask_chain<<<(B + 255) / 256, 256, 0, stream>>>(samp, head, nxt, B);
  softnp_mask_scatter<<<(B * KNN + 255) / 256, 256, 0, stream>>>(knn, samp, head,
                                                                 nxt, maskT, B);

  softnp_main<<<B / 16, 256, 0, stream>>>(zh, sqn, maskT, lossv, valv, B);
  softnp_finalize<<<1, 256, 0, stream>>>(lossv, valv, (float*)d_out, B);
}
